// GAE_LP_23536420782575
// MI455X (gfx1250) — matrix operand order repaired and checked by reading, not compiled, not run on hardware
//
#include <hip/hip_runtime.h>

// ---------------------------------------------------------------------------
// VGAE link-prediction forward for MI455X (gfx1250, wave32, WMMA + TDM).
// Outputs (flat, in order): adj [N*N], z [N*32], seq_fts [N*32].
// Workspace requirement: ~30 MB.
// ---------------------------------------------------------------------------

typedef __attribute__((ext_vector_type(16))) _Float16     v16h;
typedef __attribute__((ext_vector_type(8)))  _Float16     v8h;
typedef __attribute__((ext_vector_type(8)))  float        v8f;
typedef __attribute__((ext_vector_type(4)))  unsigned int v4u;
typedef __attribute__((ext_vector_type(8)))  int          v8i;
typedef __attribute__((ext_vector_type(4)))  int          v4i;

union HFrag { v16h v; v8h h[2]; _Float16 e[16]; };
union FAcc  { v8f  v; float e[8]; };

#define IN_DIM 256
#define HID    256
#define OUTD   32
#define KDIM   256   // K is 256 for every dense transform in this model

// fast sigmoid: one v_exp_f32 + one v_rcp_f32 (TRANS ops, co-execute w/ VALU)
__device__ __forceinline__ float fast_sigmoid(float x) {
    return __builtin_amdgcn_rcpf(1.0f + __expf(-x));
}

// ---------------------------------------------------------------- utilities
__global__ void k_zero4(float* __restrict__ p, int n4) {
    int i = blockIdx.x * blockDim.x + threadIdx.x;
    if (i < n4) ((float4*)p)[i] = make_float4(0.f, 0.f, 0.f, 0.f);
}

__global__ void k_degree(const int* __restrict__ src, const int* __restrict__ dst,
                         float* __restrict__ outdeg, float* __restrict__ indeg, int E) {
    int e = blockIdx.x * blockDim.x + threadIdx.x;
    if (e < E) {
        atomicAdd(&outdeg[src[e]], 1.0f);
        atomicAdd(&indeg[dst[e]], 1.0f);
    }
}

__global__ void k_norm(const float* __restrict__ outdeg, const float* __restrict__ indeg,
                       float* __restrict__ ns, float* __restrict__ nd, int N) {
    int i = blockIdx.x * blockDim.x + threadIdx.x;
    if (i < N) {
        ns[i] = rsqrtf(fmaxf(outdeg[i], 1.0f));
        nd[i] = rsqrtf(fmaxf(indeg[i], 1.0f));
    }
}

// ------------------------------------------------------- TDM descriptor load
// One 2D TDM transfer: tile [height x width] fp32, row stride `stride_elems`,
// from global `gptr` into LDS byte offset `lds_off` (linear x-then-y layout).
// Descriptor packing per CDNA5 ISA sec. 8.3/8.4 (D# groups 0 and 1).
// This toolchain exposes the 6-arg builtin:
//   (v4u g0, v8i g1, v4i g2, v4i g3, v8i g4, i32 cpol)
__device__ __forceinline__ void tdm_load_2d_f32(const float* gptr, unsigned lds_off,
                                                unsigned width, unsigned height,
                                                unsigned stride_elems) {
    unsigned long long ga = (unsigned long long)(size_t)gptr;
    v4u g0;
    g0[0] = 1u;                                        // count=1 (valid), user mode
    g0[1] = lds_off;                                   // lds_addr (bytes)
    g0[2] = (unsigned)(ga & 0xffffffffu);              // global_addr[31:0]
    g0[3] = (unsigned)((ga >> 32) & 0x01ffffffu)       // global_addr[56:32]
          | (2u << 30);                                // type = 2 ("image")
    v8i g1;
    g1[0] = (int)(2u << 16);                           // wg_mask=0, data_size=2 (4B)
    g1[1] = (int)((width & 0xffffu) << 16);            // tensor_dim0[15:0]
    g1[2] = (int)((width >> 16) | ((height & 0xffffu) << 16));   // td0 hi | td1 lo
    g1[3] = (int)((height >> 16) | ((width & 0xffffu) << 16));   // td1 hi | tile_dim0
    g1[4] = (int)(height & 0xffffu);                   // tile_dim1 (tile_dim2 = 0)
    g1[5] = (int)stride_elems;                         // tensor_dim0_stride[31:0]
    g1[6] = 0;                                         // stride hi | dim1_stride lo
    g1[7] = 0;                                         // dim1_stride hi
    v4i z4 = {0, 0, 0, 0};
    v8i z8 = {0, 0, 0, 0, 0, 0, 0, 0};
    __builtin_amdgcn_tensor_load_to_lds(g0, g1, z4, z4, z8, 0);
}

// ------------------------------------------------------------- generic GEMM
// C[M,16*ntiles] = (A[M,256] @ B[256, ...ldb]) * rowscale[row]  (optional)
// One wave per 16x16 tile, 8 waves/block sharing one ntile. The block's full
// 256x16 fp32 B-panel is fetched once by the Tensor Data Mover, converted to
// f16 [n][k] in LDS, then the k-loop is LDS+global(A)-only with zero barriers.
template <bool HAS_SCALE>
__global__ void k_gemm_f16wmma(const float* __restrict__ A, int lda,
                               const float* __restrict__ B, int ldb,
                               float* __restrict__ C, int ldc,
                               int M,
                               const float* __restrict__ rowscale) {
    __shared__ float    ldsBf[KDIM * 16];   // TDM landing pad: [k][n] fp32, 16KB
    __shared__ _Float16 ldsBh[16 * KDIM];   // converted: [n][k] f16, 8KB

    const int tid    = threadIdx.x;
    const int lane   = tid & 31;
    const int wave   = tid >> 5;
    const int mtiles = M >> 4;                 // multiple of 8
    const int tile   = blockIdx.x * 8 + wave;
    const int ntile  = tile / mtiles;          // identical for all waves in block
    const int mtile  = tile - ntile * mtiles;
    const int m0 = mtile << 4;
    const int n0 = ntile << 4;

    // 1) TDM: B panel [256 x 16] -> LDS (issued once, by wave 0 only)
    if (wave == 0) {
        tdm_load_2d_f32(B + n0, (unsigned)(size_t)(void*)ldsBf,
                        16u, (unsigned)KDIM, (unsigned)ldb);
        __builtin_amdgcn_s_wait_tensorcnt(0);
    }
    __syncthreads();

    // 2) one-time fp32 -> f16 transpose into [n][k] for vector ds reads
    for (int t = tid; t < KDIM * 16; t += 256) {
        int kk = t >> 4;
        int nn = t & 15;
        ldsBh[nn * KDIM + kk] = (_Float16)ldsBf[t];
    }
    __syncthreads();

    const int lm  = lane & 15;
    const int hi  = lane >> 4;
    const int kbA = hi ? 8 : 0;                // A lane-half K base (ISA 7.12.2)
    const int kbB = hi ? 8 : 0;                // B lane-half K base

    FAcc acc;
#pragma unroll
    for (int j = 0; j < 8; ++j) acc.e[j] = 0.0f;

    const float*    arow = A + (m0 + lm) * lda;
    const _Float16* brow = ldsBh + lm * KDIM;

#pragma unroll 4
    for (int k0 = 0; k0 < KDIM; k0 += 32) {
        HFrag a, b;
        const float* ap = arow + k0 + kbA;
#pragma unroll
        for (int j = 0; j < 8; ++j) a.e[j]     = (_Float16)ap[j];        // K=kbA+0..7
#pragma unroll
        for (int j = 0; j < 8; ++j) a.e[8 + j] = (_Float16)ap[16 + j];   // K=kbA+16..23

        const _Float16* bp = brow + k0 + kbB;                            // K=kbB+0..7, kbB+16..23
        b.h[0] = *(const v8h*)bp;
        b.h[1] = *(const v8h*)(bp + 16);

        acc.v = __builtin_amdgcn_wmma_f32_16x16x32_f16(
            false, a.v, false, b.v, (short)0, acc.v, false, false);
    }

#pragma unroll
    for (int vv = 0; vv < 8; ++vv) {
        int row = m0 + vv + (hi << 3);         // C layout: M = vgpr + 8*lanehi
        float out = acc.e[vv];
        if (HAS_SCALE) out *= rowscale[row];
        C[row * ldc + n0 + lm] = out;
    }
}

// --------------------------------------------------------- edge aggregation
__global__ void k_edge_agg256(const float* __restrict__ T, const int* __restrict__ src,
                              const int* __restrict__ dst, float* __restrict__ agg, int E) {
    int idx = blockIdx.x * blockDim.x + threadIdx.x;     // E * 64 threads
    if (idx >= E * 64) return;
    int e = idx >> 6;
    int c = (idx & 63) << 2;
    __builtin_prefetch(src + ((e + 4096 < E) ? e + 4096 : e), 0, 0);
    int s = src[e], d = dst[e];
    const float4 vv = *(const float4*)(T + (long)s * 256 + c);
    float* o = agg + (long)d * 256 + c;
    atomicAdd(o + 0, vv.x); atomicAdd(o + 1, vv.y);
    atomicAdd(o + 2, vv.z); atomicAdd(o + 3, vv.w);
}

__global__ void k_edge_agg32x2(const float* __restrict__ T2, const float* __restrict__ T3,
                               const int* __restrict__ src, const int* __restrict__ dst,
                               float* __restrict__ agg2, float* __restrict__ agg3, int E) {
    int idx = blockIdx.x * blockDim.x + threadIdx.x;     // E * 8 threads
    if (idx >= E * 8) return;
    int e = idx >> 3;
    int c = (idx & 7) << 2;
    int s = src[e], d = dst[e];
    const float4 va = *(const float4*)(T2 + (long)s * 32 + c);
    const float4 vb = *(const float4*)(T3 + (long)s * 32 + c);
    float* o2 = agg2 + (long)d * 32 + c;
    float* o3 = agg3 + (long)d * 32 + c;
    atomicAdd(o2 + 0, va.x); atomicAdd(o2 + 1, va.y);
    atomicAdd(o2 + 2, va.z); atomicAdd(o2 + 3, va.w);
    atomicAdd(o3 + 0, vb.x); atomicAdd(o3 + 1, vb.y);
    atomicAdd(o3 + 2, vb.z); atomicAdd(o3 + 3, vb.w);
}

// h = relu(agg * norm_dst[row] + b[col])
__global__ void k_bias_relu(const float* __restrict__ agg, const float* __restrict__ nd,
                            const float* __restrict__ b, float* __restrict__ h, int total, int F) {
    int idx = blockIdx.x * blockDim.x + threadIdx.x;
    if (idx >= total) return;
    int r = idx / F;
    int c = idx - r * F;
    h[idx] = fmaxf(agg[idx] * nd[r] + b[c], 0.0f);
}

// z = relu(agg2*nd+b2) + noise * exp(relu(agg3*nd+b3)); emit f32 + f16 copies
__global__ void k_make_z(const float* __restrict__ agg2, const float* __restrict__ agg3,
                         const float* __restrict__ nd, const float* __restrict__ b2,
                         const float* __restrict__ b3, const float* __restrict__ noise,
                         float* __restrict__ zout, _Float16* __restrict__ zh, int total) {
    int idx = blockIdx.x * blockDim.x + threadIdx.x;
    if (idx >= total) return;
    int i = idx >> 5;
    int c = idx & 31;
    float ndv  = nd[i];
    float mean = fmaxf(agg2[idx] * ndv + b2[c], 0.0f);
    float ls   = fmaxf(agg3[idx] * ndv + b3[c], 0.0f);
    float z    = mean + noise[idx] * __expf(ls);
    zout[idx] = z;
    zh[idx]   = (_Float16)z;
}

// adj = sigmoid(z @ z^T): K = 32 == exactly one WMMA per 16x16 output tile.
// 268 MB result is written once and never re-read -> non-temporal stores.
__global__ void k_zzt_sigmoid(const _Float16* __restrict__ zh, float* __restrict__ adj, int N) {
    const int lane   = threadIdx.x & 31;
    const int wave   = threadIdx.x >> 5;
    const int mtiles = N >> 4;
    const int tile   = blockIdx.x * 8 + wave;
    const int ntile  = tile / mtiles;
    const int mtile  = tile - ntile * mtiles;
    const int m0 = mtile << 4;
    const int n0 = ntile << 4;
    const int lm = lane & 15;
    const int hi = lane >> 4;

    HFrag a, b;
    const _Float16* ap = zh + (m0 + lm) * 32 + (hi ? 8 : 0);
    a.h[0] = *(const v8h*)ap;            // K = kbA+0..7
    a.h[1] = *(const v8h*)(ap + 16);     // K = kbA+16..23
    const _Float16* bp = zh + (n0 + lm) * 32 + (hi ? 8 : 0);   // B[k][n] = z[n][k]
    b.h[0] = *(const v8h*)bp;            // K = kbB+0..7
    b.h[1] = *(const v8h*)(bp + 16);     // K = kbB+16..23

    FAcc acc;
#pragma unroll
    for (int j = 0; j < 8; ++j) acc.e[j] = 0.0f;
    acc.v = __builtin_amdgcn_wmma_f32_16x16x32_f16(
        false, a.v, false, b.v, (short)0, acc.v, false, false);

#pragma unroll
    for (int vv = 0; vv < 8; ++vv) {
        int row = m0 + vv + (hi << 3);
        __builtin_nontemporal_store(fast_sigmoid(acc.e[vv]),
                                    adj + (long)row * N + n0 + lm);
    }
}

// ---------------------------------------------------------------------------
extern "C" void kernel_launch(void* const* d_in, const int* in_sizes, int n_in,
                              void* d_out, int out_size, void* d_ws, size_t ws_size,
                              hipStream_t stream) {
    const float* features = (const float*)d_in[0];
    const float* noise    = (const float*)d_in[1];
    const int*   src      = (const int*)d_in[2];
    const int*   dst      = (const int*)d_in[3];
    const float* W1       = (const float*)d_in[4];
    const float* b1       = (const float*)d_in[5];
    const float* W2       = (const float*)d_in[6];
    const float* b2       = (const float*)d_in[7];
    const float* W3       = (const float*)d_in[8];
    const float* b3       = (const float*)d_in[9];
    const float* fc_w     = (const float*)d_in[10];

    const int N = in_sizes[0] / IN_DIM;     // 8192
    const int E = in_sizes[2];              // 262144

    // ---- workspace layout (floats); zeroed region first ----
    float* ws = (float*)d_ws;
    size_t o = 0;
    float* outdeg = ws + o; o += N;
    float* indeg  = ws + o; o += N;
    float* agg1   = ws + o; o += (size_t)N * HID;
    float* agg2   = ws + o; o += (size_t)N * OUTD;
    float* agg3   = ws + o; o += (size_t)N * OUTD;
    const size_t zero_floats = o;           // everything above must start at 0
    float* norm_s = ws + o; o += N;
    float* norm_d = ws + o; o += N;
    float* T1     = ws + o; o += (size_t)N * HID;   // (X@W1) * norm_s
    float* h      = ws + o; o += (size_t)N * HID;
    float* T2s    = ws + o; o += (size_t)N * OUTD;  // (h@W2[:, :32]) * norm_s
    float* T3s    = ws + o; o += (size_t)N * OUTD;  // (h@W3) * norm_s
    _Float16* zh  = (_Float16*)(ws + o);            // [N, 32] f16 z copy

    // ---- output layout ----
    float* adj  = (float*)d_out;                      // [N, N]
    float* zout = adj + (size_t)N * N;                // [N, 32]
    float* seq  = zout + (size_t)N * OUTD;            // [N, 32]

    const int BLK = 256;

    // 1) zero accumulators
    k_zero4<<<(int)((zero_floats / 4 + BLK - 1) / BLK), BLK, 0, stream>>>(ws, (int)(zero_floats / 4));

    // 2) degrees + norms
    k_degree<<<(E + BLK - 1) / BLK, BLK, 0, stream>>>(src, dst, outdeg, indeg, E);
    k_norm<<<(N + BLK - 1) / BLK, BLK, 0, stream>>>(outdeg, indeg, norm_s, norm_d, N);

    // 3) T1 = (features @ W1) * norm_s
    k_gemm_f16wmma<true><<<(N / 16) * (HID / 16) / 8, BLK, 0, stream>>>(
        features, IN_DIM, W1, HID, T1, HID, N, norm_s);

    // 4) agg1 = segment_sum(T1[src] -> dst)
    k_edge_agg256<<<(E * 64) / BLK, BLK, 0, stream>>>(T1, src, dst, agg1, E);

    // 5) h = relu(agg1 * norm_d + b1)
    k_bias_relu<<<(N * HID) / BLK, BLK, 0, stream>>>(agg1, norm_d, b1, h, N * HID, HID);

    // 6) second-layer transforms (mean needs only first 32 of 256 cols of W2)
    k_gemm_f16wmma<true><<<(N / 16) * (OUTD / 16) / 8, BLK, 0, stream>>>(
        h, HID, W2, HID, T2s, OUTD, N, norm_s);       // W2[:, :32] via ldb=HID
    k_gemm_f16wmma<true><<<(N / 16) * (OUTD / 16) / 8, BLK, 0, stream>>>(
        h, HID, W3, OUTD, T3s, OUTD, N, norm_s);

    // 7) agg2/agg3 segment sums (32-wide, both in one pass)
    k_edge_agg32x2<<<(E * 8) / BLK, BLK, 0, stream>>>(T2s, T3s, src, dst, agg2, agg3, E);

    // 8) z
    k_make_z<<<(N * OUTD) / BLK, BLK, 0, stream>>>(agg2, agg3, norm_d, b2, b3, noise,
                                                   zout, zh, N * OUTD);

    // 9) seq_fts = features @ fc_w (no row scale)
    k_gemm_f16wmma<false><<<(N / 16) * (OUTD / 16) / 8, BLK, 0, stream>>>(
        features, IN_DIM, fc_w, OUTD, seq, OUTD, N, nullptr);

    // 10) adj = sigmoid(z @ z^T)  — store-bound 268 MB, one WMMA per tile
    k_zzt_sigmoid<<<(N / 16) * (N / 16) / 8, BLK, 0, stream>>>(zh, adj, N);
}
